// HOPEAttentionBlock_12893491823311
// MI455X (gfx1250) — compile-verified, matrix operand order checked by reading, not run on hardware
//
#include <hip/hip_runtime.h>
#include <hip/hip_bf16.h>

// ---------------------------------------------------------------------------
// CDNA5 (gfx1250) HOPE block: flash attention + CMS MLPs + TTT scans.
// - v_wmma_f32_16x16x32_bf16 everywhere (wave32, 16x16 C tiles, f32 accum)
// - Tensor Data Mover (tensor_load_to_lds) for bf16 A tiles -> LDS,
//   double-buffered and overlapped with WMMA of the previous k-tile
// - bf16 activation mirrors so TDM moves raw bytes; fp32 kept where needed
// - 16B vectorized LDS/global fragment loads (A-fragment = two 8x bf16 runs)
// - branch-free staging via index clamping (OOB tiles feed discarded C elems)
// ---------------------------------------------------------------------------

typedef __attribute__((ext_vector_type(16))) __bf16 v16bf;
typedef __attribute__((ext_vector_type(8)))  __bf16 v8bf;
typedef __attribute__((ext_vector_type(8)))  float  v8f;
typedef unsigned int u32x4 __attribute__((ext_vector_type(4)));
typedef int          i32x4 __attribute__((ext_vector_type(4)));
typedef int          i32x8 __attribute__((ext_vector_type(8)));

union Frag {
    v16bf v;
    v8bf  h[2];
};
union PackBF2 {
    unsigned u;
    __bf16   h[2];
};

#if __has_builtin(__builtin_amdgcn_tensor_load_to_lds) && \
    __has_builtin(__builtin_amdgcn_s_wait_tensorcnt)
#define HAVE_TDM 1
#else
#define HAVE_TDM 0
#endif

__device__ __forceinline__ float gelu_f(float x) {
    float x3 = x * x * x;
    float u  = 0.7978845608028654f * (x + 0.044715f * x3);
    return 0.5f * x * (1.f + tanhf(u));
}
__device__ __forceinline__ float dgelu_f(float x) {
    float x2 = x * x;
    float u  = 0.7978845608028654f * (x + 0.044715f * x * x2);
    float t  = tanhf(u);
    float du = 0.7978845608028654f * (1.f + 3.f * 0.044715f * x2);
    return 0.5f * (1.f + t) + 0.5f * x * (1.f - t * t) * du;
}

#if HAVE_TDM
// Issue one 2D TDM tile load: tensor row-major [d1 x d0] bf16, tile
// [tile_d1 x tile_d0], dense into LDS at lds_off. D# per ISA 8.3/8.4.
__device__ __forceinline__ void tdm_load_2d(unsigned lds_off,
                                            const __bf16* gptr, int tens_d0,
                                            int tens_d1, int tile_d0,
                                            int tile_d1, long long stride0) {
    unsigned long long ga = (unsigned long long)(size_t)gptr;
    u32x4 g0;
    g0[0] = 1u;  // count=1, user descriptor, no gather
    g0[1] = lds_off;
    g0[2] = (unsigned)(ga & 0xFFFFFFFFu);
    g0[3] = (unsigned)((ga >> 32) & 0x01FFFFFFu) | 0x80000000u;  // type=2
    i32x8 g1;
    g1[0] = 0x00010000;  // data_size = 1 (2 bytes/elem)
    g1[1] = (int)(((unsigned)tens_d0 & 0xFFFFu) << 16);
    g1[2] = (int)((((unsigned)tens_d0 >> 16) & 0xFFFFu) |
                  (((unsigned)tens_d1 & 0xFFFFu) << 16));
    g1[3] = (int)((((unsigned)tens_d1 >> 16) & 0xFFFFu) |
                  ((unsigned)tile_d0 << 16));
    g1[4] = tile_d1 & 0xFFFF;  // tile_dim1; tile_dim2 = 0
    g1[5] = (int)((unsigned long long)stride0 & 0xFFFFFFFFull);
    g1[6] = (int)(((unsigned long long)stride0 >> 32) & 0xFFFFull);
    g1[7] = 0;
    i32x4 z4 = {0, 0, 0, 0};
#if defined(__clang_major__) && (__clang_major__ >= 23)
    i32x8 z8 = {0, 0, 0, 0, 0, 0, 0, 0};
    __builtin_amdgcn_tensor_load_to_lds(g0, g1, z4, z4, z8, 0);
#else
    __builtin_amdgcn_tensor_load_to_lds(g0, g1, z4, z4, 0);
#endif
}
#endif

// ---------------------------------------------------------------------------
// GEMM (TDM variant): C[M,N] = A[M,K](bf16) @ op(B)[K,N](f32) (+bias)(+epi)
// Double-buffered pipeline: TDM DMA of A tile t+1 + B staging of tile t+1
// overlap the WMMAs of tile t. ACT: 0 none, 1 gelu, 2 gelu + store z (f32).
// EPI: 0 store, 1 dpred = scal[0]*(pred-tgt)*rowmask[m], 2 dz = acc*dgelu(z).
// Outputs: Cf (f32, optional) and Ch (bf16, optional).
// ---------------------------------------------------------------------------
template <int HAS_BIAS, int ACT, int EPI, int TRANSB>
__global__ __launch_bounds__(256) void gemm_tdm(
    const __bf16* __restrict__ A, const float* __restrict__ B,
    float* __restrict__ Cf, __bf16* __restrict__ Ch, int M, int N, int K,
    const float* __restrict__ bias, float* __restrict__ zbuf,
    const float* __restrict__ tgt, const float* __restrict__ rowmask,
    const float* __restrict__ scal) {
    __shared__ __bf16 As[2][64][32];    // dense: TDM target, 64B row pitch
    __shared__ __bf16 Bst[2][128][40];  // transposed B, 80B pitch

    const int tid  = threadIdx.x;
    const int lane = tid & 31, wave = tid >> 5;
    const int mt = wave & 3, ng = wave >> 2;
    const int lm = lane & 15, hi = lane >> 4;
    const int m0 = blockIdx.y * 64, n0 = blockIdx.x * 128;

    v8f acc[4];
    for (int t = 0; t < 4; ++t)
        for (int i = 0; i < 8; ++i) acc[t][i] = 0.f;

#if HAVE_TDM
    const unsigned ldsA0 = (unsigned)(size_t)(&As[0][0][0]);
#endif

    auto issueA = [&](int buf, int k0) {
#if HAVE_TDM
        if (wave == 0)
            tdm_load_2d(ldsA0 + (unsigned)buf * (64 * 32 * 2),
                        A + (size_t)m0 * K + k0, K - k0, M - m0, 32, 64,
                        (long long)K);
#else
        for (int i = tid; i < 64 * 32; i += 256) {
            int m = i >> 5, kk = i & 31;
            int gm = m0 + m;
            gm            = gm < M ? gm : M - 1;  // clamp: OOB rows discarded
            As[buf][m][kk] = A[(size_t)gm * K + k0 + kk];
        }
#endif
    };
    auto waitA = [&]() {
#if HAVE_TDM
        if (wave == 0) __builtin_amdgcn_s_wait_tensorcnt(0);
#endif
    };
    // stage B tile 32x128 (f32 -> bf16), two k-consecutive elems per thread,
    // packed 4B LDS stores into the transposed layout. Branch-free (clamped).
    auto stageB = [&](int buf, int k0) {
        for (int i = tid; i < 16 * 128; i += 256) {
            int kk = (i >> 7) << 1, n = i & 127;
            int gk = k0 + kk;
            int gn = n0 + n;
            gn = gn < N ? gn : N - 1;  // clamp: OOB cols feed discarded C
            float v0, v1;
            if (TRANSB) {
                const float* p = B + (size_t)gn * K + gk;
                v0 = p[0];
                v1 = p[1];
            } else {
                v0 = B[(size_t)gk * N + gn];
                v1 = B[(size_t)(gk + 1) * N + gn];
            }
            PackBF2 pk;
            pk.h[0] = (__bf16)v0;
            pk.h[1] = (__bf16)v1;
            *(unsigned*)&Bst[buf][n][kk] = pk.u;
        }
    };

    // pipeline prologue
    issueA(0, 0);
    stageB(0, 0);
    waitA();
    __syncthreads();

    const int nk = K >> 5;  // K is a multiple of 32 in all uses
    for (int t = 0; t < nk; ++t) {
        const int cur = t & 1;
        const int k0n = (t + 1) << 5;
        if (k0n < K) {  // kick off tile t+1 while computing tile t
            issueA(cur ^ 1, k0n);
            stageB(cur ^ 1, k0n);
            const int k0p = (t + 2) << 5;  // prefetch tile t+2 into GL2
            if (k0p < K) {
                int pn = n0 + (tid & 127);
                if (pn < N) {
                    const float* pf = TRANSB ? B + (size_t)pn * K + k0p
                                             : B + (size_t)k0p * N + pn;
                    __builtin_prefetch(pf, 0, 1);
                }
            }
        }
        Frag af;
        af.h[0] = *(const v8bf*)&As[cur][mt * 16 + lm][8 * hi];
        af.h[1] = *(const v8bf*)&As[cur][mt * 16 + lm][16 + 8 * hi];
#pragma unroll
        for (int t4 = 0; t4 < 4; ++t4) {
            Frag bf;
            const int col = ng * 64 + t4 * 16 + lm;
            bf.h[0]       = *(const v8bf*)&Bst[cur][col][8 * hi];
            bf.h[1]       = *(const v8bf*)&Bst[cur][col][16 + 8 * hi];
            acc[t4] = __builtin_amdgcn_wmma_f32_16x16x32_bf16(
                false, af.v, false, bf.v, (short)0, acc[t4], false, false);
        }
        if (k0n < K) waitA();
        __syncthreads();
    }

#pragma unroll
    for (int t = 0; t < 4; ++t) {
#pragma unroll
        for (int r = 0; r < 8; ++r) {
            int gm = m0 + mt * 16 + r + 8 * hi;
            int gn = n0 + ng * 64 + t * 16 + lm;
            if (gm < M && gn < N) {
                float  x   = acc[t][r];
                size_t idx = (size_t)gm * N + gn;
                if (HAS_BIAS) x += bias[gn];
                float val;
                if (ACT == 1) {
                    val = gelu_f(x);
                } else if (ACT == 2) {
                    zbuf[idx] = x;
                    val       = gelu_f(x);
                } else if (EPI == 1) {
                    val = scal[0] * (x - tgt[idx]) * rowmask[gm];
                } else if (EPI == 2) {
                    val = x * dgelu_f(zbuf[idx]);
                } else {
                    val = x;
                }
                if (Cf) Cf[idx] = val;
                if (Ch) Ch[idx] = (__bf16)val;
            }
        }
    }
}

// ---------------------------------------------------------------------------
// Weight update GEMM: W[M,N] -= lr*scal[1]*(A^T @ B), A bf16 [K,M],
// B bf16 [K,N], K = chunk rows (32/128). Branch-free clamped staging.
// ---------------------------------------------------------------------------
__global__ __launch_bounds__(256) void gemm_wupd(
    const __bf16* __restrict__ A, const __bf16* __restrict__ B,
    float* __restrict__ W, int M, int N, int K,
    const float* __restrict__ scal, float lr) {
    __shared__ __bf16 As[64][32];
    __shared__ __bf16 Bst[128][40];
    const int tid  = threadIdx.x;
    const int lane = tid & 31, wave = tid >> 5;
    const int mt = wave & 3, ng = wave >> 2;
    const int lm = lane & 15, hi = lane >> 4;
    const int m0 = blockIdx.y * 64, n0 = blockIdx.x * 128;

    v8f acc[4];
    for (int t = 0; t < 4; ++t)
        for (int i = 0; i < 8; ++i) acc[t][i] = 0.f;

    for (int k0 = 0; k0 < K; k0 += 32) {
        for (int i = tid; i < 64 * 32; i += 256) {
            int m = i >> 5, kk = i & 31;
            int gm = m0 + m;
            gm        = gm < M ? gm : M - 1;
            As[m][kk] = A[(size_t)(k0 + kk) * M + gm];
        }
        for (int i = tid; i < 16 * 128; i += 256) {
            int kk = (i >> 7) << 1, n = i & 127;
            int gk = k0 + kk;
            int gn = n0 + n;
            gn = gn < N ? gn : N - 1;
            PackBF2 pk;
            pk.h[0] = B[(size_t)gk * N + gn];
            pk.h[1] = B[(size_t)(gk + 1) * N + gn];
            *(unsigned*)&Bst[n][kk] = pk.u;
        }
        __syncthreads();
        Frag af;
        af.h[0] = *(const v8bf*)&As[mt * 16 + lm][8 * hi];
        af.h[1] = *(const v8bf*)&As[mt * 16 + lm][16 + 8 * hi];
#pragma unroll
        for (int t = 0; t < 4; ++t) {
            Frag bf;
            const int col = ng * 64 + t * 16 + lm;
            bf.h[0]       = *(const v8bf*)&Bst[col][8 * hi];
            bf.h[1]       = *(const v8bf*)&Bst[col][16 + 8 * hi];
            acc[t] = __builtin_amdgcn_wmma_f32_16x16x32_bf16(
                false, af.v, false, bf.v, (short)0, acc[t], false, false);
        }
        __syncthreads();
    }
    const float s = lr * scal[1];
#pragma unroll
    for (int t = 0; t < 4; ++t) {
#pragma unroll
        for (int r = 0; r < 8; ++r) {
            int gm = m0 + mt * 16 + r + 8 * hi;
            int gn = n0 + ng * 64 + t * 16 + lm;
            if (gm < M && gn < N) W[(size_t)gm * N + gn] -= s * acc[t][r];
        }
    }
}

// ---------------------------------------------------------------------------
// Flash attention (causal), bf16 q/k/v in, bf16 out (pre-Wo).
// grid = (S/64, H, B), block = 128 (4 waves), 16 q rows/wave, Dh=64.
// ---------------------------------------------------------------------------
__global__ __launch_bounds__(128) void flash_attn(
    const __bf16* __restrict__ q, const __bf16* __restrict__ k,
    const __bf16* __restrict__ v, __bf16* __restrict__ out, int Bn, int S,
    int H) {
    const int D = 1024;
    const int b = blockIdx.z, h = blockIdx.y;
    const int qblk = blockIdx.x * 64;
    const int wave = threadIdx.x >> 5, lane = threadIdx.x & 31;
    const int lm = lane & 15, hi = lane >> 4;
    const int qb = qblk + wave * 16;

    __shared__ __bf16 Ps[4][16][40];  // 80B pitch (16B aligned rows)

    Frag aq[2];
    {
        const __bf16* qrow = q + ((size_t)(b * S) + qb + lm) * D + h * 64;
#pragma unroll
        for (int c = 0; c < 2; ++c) {
            aq[c].h[0] = *(const v8bf*)(qrow + c * 32 + 8 * hi);
            aq[c].h[1] = *(const v8bf*)(qrow + c * 32 + 16 + 8 * hi);
        }
    }

    v8f   o[4];
    float mrow[8], lrow[8];
    for (int t = 0; t < 4; ++t)
        for (int i = 0; i < 8; ++i) o[t][i] = 0.f;
    for (int i = 0; i < 8; ++i) { mrow[i] = -3.0e38f; lrow[i] = 0.f; }

    const int kvmax = qblk + 63;  // uniform per block (causal)
    for (int kv0 = 0; kv0 <= kvmax; kv0 += 32) {
        v8f sc[2];
#pragma unroll
        for (int g = 0; g < 2; ++g) {
            for (int i = 0; i < 8; ++i) sc[g][i] = 0.f;
            const __bf16* krow =
                k + ((size_t)(b * S) + kv0 + g * 16 + lm) * D + h * 64;
#pragma unroll
            for (int c = 0; c < 2; ++c) {
                Frag bk;
                bk.h[0] = *(const v8bf*)(krow + c * 32 + 8 * hi);
                bk.h[1] = *(const v8bf*)(krow + c * 32 + 16 + 8 * hi);
                sc[g]   = __builtin_amdgcn_wmma_f32_16x16x32_bf16(
                    false, aq[c].v, false, bk.v, (short)0, sc[g], false,
                    false);
            }
        }
#pragma unroll
        for (int r = 0; r < 8; ++r) {
            int   qr = qb + r + 8 * hi;
            float s0 = sc[0][r] * 0.125f, s1 = sc[1][r] * 0.125f;
            if (kv0 + lm > qr) s0 = -3.0e30f;
            if (kv0 + 16 + lm > qr) s1 = -3.0e30f;
            float mx = fmaxf(s0, s1);
            for (int off = 1; off < 16; off <<= 1)
                mx = fmaxf(mx, __shfl_xor(mx, off, 32));
            float mnew  = fmaxf(mrow[r], mx);
            float alpha = expf(mrow[r] - mnew);
            float p0 = expf(s0 - mnew), p1 = expf(s1 - mnew);
            float ps = p0 + p1;
            for (int off = 1; off < 16; off <<= 1)
                ps += __shfl_xor(ps, off, 32);
            lrow[r] = lrow[r] * alpha + ps;
            mrow[r] = mnew;
#pragma unroll
            for (int t = 0; t < 4; ++t) o[t][r] *= alpha;
            Ps[wave][r + 8 * hi][lm]      = (__bf16)p0;
            Ps[wave][r + 8 * hi][16 + lm] = (__bf16)p1;
        }
        __syncthreads();
        Frag ap;
        ap.h[0] = *(const v8bf*)&Ps[wave][lm][8 * hi];
        ap.h[1] = *(const v8bf*)&Ps[wave][lm][16 + 8 * hi];
#pragma unroll
        for (int t = 0; t < 4; ++t) {
            Frag bv;
#pragma unroll
            for (int j = 0; j < 8; ++j) {
                int k0a = j + 8 * hi;       // frag run 0: K = 8*hi + j
                int k1a = 16 + 8 * hi + j;  // frag run 1: K = 16 + 8*hi + j
                bv.v[j] = v[((size_t)(b * S) + kv0 + k0a) * D + h * 64 +
                            t * 16 + lm];
                bv.v[j + 8] = v[((size_t)(b * S) + kv0 + k1a) * D + h * 64 +
                                t * 16 + lm];
            }
            o[t] = __builtin_amdgcn_wmma_f32_16x16x32_bf16(
                false, ap.v, false, bv.v, (short)0, o[t], false, false);
        }
        __syncthreads();
    }
#pragma unroll
    for (int r = 0; r < 8; ++r) {
        int   qr  = qb + r + 8 * hi;
        float inv = 1.f / lrow[r];
#pragma unroll
        for (int t = 0; t < 4; ++t)
            out[((size_t)(b * S) + qr) * D + h * 64 + t * 16 + lm] =
                (__bf16)(o[t][r] * inv);
    }
}

// ---------------------------------------------------------------------------
__global__ __launch_bounds__(256) void f32_to_bf16(const float* __restrict__ s,
                                                   __bf16* __restrict__ d,
                                                   int n) {
    int i = blockIdx.x * 256 + threadIdx.x;
    if (i < n) d[i] = (__bf16)s[i];
}

// TTT chunk prep: active mask, per-batch delta, tgt (f32), ci copies (f32 +
// bf16), scal[0]=2/max(total,1), scal[1]=any_active. grid(1), block 256.
__global__ __launch_bounds__(256) void ttt_prep(
    const float* __restrict__ lin, const float* __restrict__ teach, int S,
    int D, int P, int c0, int Bn, float* __restrict__ cic,
    __bf16* __restrict__ cic16, float* __restrict__ tgt,
    float* __restrict__ rowmask, float* __restrict__ scal) {
    __shared__ float red[256];
    __shared__ float act[128];
    __shared__ float denomb[8];
    const int tid = threadIdx.x;
    const int R   = Bn * P;

    for (int rr = 0; rr < R; ++rr) {
        int          bb = rr / P, p = rr % P;
        const float* tr = teach + ((size_t)bb * S + c0 + p) * D;
        float        s  = 0.f;
        for (int d = tid; d < D; d += 256) s += fabsf(tr[d]);
        red[tid] = s;
        __syncthreads();
        for (int st = 128; st > 0; st >>= 1) {
            if (tid < st) red[tid] += red[tid + st];
            __syncthreads();
        }
        if (tid == 0) act[rr] = (red[0] > 0.f) ? 1.f : 0.f;
        __syncthreads();
    }
    if (tid == 0) {
        float total = 0.f;
        for (int bb = 0; bb < Bn; ++bb) {
            float cb = 0.f;
            for (int p = 0; p < P; ++p) cb += act[bb * P + p];
            denomb[bb] = fmaxf(cb, 1.f);
            total += cb;
        }
        scal[0] = 2.f / fmaxf(total, 1.f);
        scal[1] = (total > 0.f) ? 1.f : 0.f;
    }
    __syncthreads();
    for (int rr = tid; rr < R; rr += 256) rowmask[rr] = act[rr];
    for (int d = tid; d < D; d += 256) {
        for (int bb = 0; bb < Bn; ++bb) {
            float s = 0.f;
            for (int p = 0; p < P; ++p)
                s += act[bb * P + p] * teach[((size_t)bb * S + c0 + p) * D + d];
            float delta = s / denomb[bb];
            for (int p = 0; p < P; ++p) {
                int    rr  = bb * P + p;
                float  ci  = lin[((size_t)bb * S + c0 + p) * D + d];
                size_t idx = (size_t)rr * D + d;
                cic[idx]   = ci;
                cic16[idx] = (__bf16)ci;
                tgt[idx]   = ci + delta;
            }
        }
    }
}

__global__ __launch_bounds__(256) void bias_update(
    float* __restrict__ bvec, const __bf16* __restrict__ G, int R, int Ncols,
    const float* __restrict__ scal, float lr) {
    int n = blockIdx.x * 256 + threadIdx.x;
    if (n >= Ncols) return;
    float s = 0.f;
    for (int r = 0; r < R; ++r) s += (float)G[(size_t)r * Ncols + n];
    bvec[n] -= lr * scal[1] * s;
}

// ---------------------------------------------------------------------------
extern "C" void kernel_launch(void* const* d_in, const int* in_sizes, int n_in,
                              void* d_out, int out_size, void* d_ws,
                              size_t ws_size, hipStream_t stream) {
    (void)in_sizes; (void)n_in; (void)out_size; (void)ws_size;
    const int   Bn = 2, S = 2048, D = 1024, H = 16, HID = 4096;
    const float LR = 1e-3f;
    const int   BS = Bn * S;             // 4096 rows
    const size_t NACT = (size_t)BS * D;  // 4,194,304 elems

    const float* x      = (const float*)d_in[0];
    const float* te     = (const float*)d_in[1];
    const float* wq     = (const float*)d_in[2];
    const float* wk     = (const float*)d_in[3];
    const float* wv     = (const float*)d_in[4];
    const float* wo     = (const float*)d_in[5];
    const float* w1f_in = (const float*)d_in[6];
    const float* b1f_in = (const float*)d_in[7];
    const float* w2f_in = (const float*)d_in[8];
    const float* b2f_in = (const float*)d_in[9];
    const float* w1s_in = (const float*)d_in[10];
    const float* b1s_in = (const float*)d_in[11];
    const float* w2s_in = (const float*)d_in[12];
    const float* b2s_in = (const float*)d_in[13];

    // workspace (256B-aligned bump allocator)
    char* cur   = (char*)d_ws;
    auto  alloc = [&](size_t bytes) {
        char* p = cur;
        cur += (bytes + 255) & ~(size_t)255;
        return p;
    };
    __bf16* regA     = (__bf16*)alloc(4 * NACT * 2);  // q|k|v|ao, reused hbig16
    __bf16* qb16     = regA;
    __bf16* kb16     = regA + NACT;
    __bf16* vb16     = regA + 2 * NACT;
    __bf16* ao16     = regA + 3 * NACT;
    __bf16* hbig16   = regA;  // [BS, HID] bf16
    float*  infast   = (float*)alloc(NACT * 4);
    float*  inslow   = (float*)alloc(NACT * 4);
    __bf16* x16      = (__bf16*)alloc(NACT * 2);
    __bf16* infast16 = (__bf16*)alloc(NACT * 2);
    __bf16* inslow16 = (__bf16*)alloc(NACT * 2);
    float*  cic      = (float*)alloc(131072 * 4);
    __bf16* cic16    = (__bf16*)alloc(131072 * 2);
    float*  tgt      = (float*)alloc(131072 * 4);
    float*  zb       = (float*)alloc(524288 * 4);
    __bf16* hb16     = (__bf16*)alloc(524288 * 2);
    __bf16* dpred16  = (__bf16*)alloc(131072 * 2);
    __bf16* dz16     = (__bf16*)alloc(524288 * 2);
    float*  rmask    = (float*)alloc(128 * 4);
    float*  scal     = (float*)alloc(64);

    float* out = (float*)d_out;
    float* cms = out;
    float* w1f = out + NACT;
    float* b1f = w1f + 4194304;
    float* w2f = b1f + 4096;
    float* b2f = w2f + 4194304;
    float* w1s = b2f + 1024;
    float* b1s = w1s + 4194304;
    float* w2s = b1s + 4096;
    float* b2s = w2s + 4194304;

    // seed output params (scan carry lives in d_out, updated in place)
    hipMemcpyAsync(w1f, w1f_in, 4194304 * 4, hipMemcpyDeviceToDevice, stream);
    hipMemcpyAsync(b1f, b1f_in, 4096 * 4, hipMemcpyDeviceToDevice, stream);
    hipMemcpyAsync(w2f, w2f_in, 4194304 * 4, hipMemcpyDeviceToDevice, stream);
    hipMemcpyAsync(b2f, b2f_in, 1024 * 4, hipMemcpyDeviceToDevice, stream);
    hipMemcpyAsync(w1s, w1s_in, 4194304 * 4, hipMemcpyDeviceToDevice, stream);
    hipMemcpyAsync(b1s, b1s_in, 4096 * 4, hipMemcpyDeviceToDevice, stream);
    hipMemcpyAsync(w2s, w2s_in, 4194304 * 4, hipMemcpyDeviceToDevice, stream);
    hipMemcpyAsync(b2s, b2s_in, 1024 * 4, hipMemcpyDeviceToDevice, stream);

    dim3 blk(256);
    auto mkgrid = [](int M, int N) {
        return dim3((unsigned)((N + 127) / 128), (unsigned)((M + 63) / 64));
    };

    // --- attention ---
    f32_to_bf16<<<dim3((unsigned)((NACT + 255) / 256)), blk, 0, stream>>>(
        x, x16, (int)NACT);
    gemm_tdm<0, 0, 0, 0><<<mkgrid(BS, D), blk, 0, stream>>>(
        x16, wq, nullptr, qb16, BS, D, D, nullptr, nullptr, nullptr, nullptr,
        nullptr);
    gemm_tdm<0, 0, 0, 0><<<mkgrid(BS, D), blk, 0, stream>>>(
        x16, wk, nullptr, kb16, BS, D, D, nullptr, nullptr, nullptr, nullptr,
        nullptr);
    gemm_tdm<0, 0, 0, 0><<<mkgrid(BS, D), blk, 0, stream>>>(
        x16, wv, nullptr, vb16, BS, D, D, nullptr, nullptr, nullptr, nullptr,
        nullptr);
    flash_attn<<<dim3(S / 64, H, Bn), dim3(128), 0, stream>>>(qb16, kb16,
                                                              vb16, ao16, Bn,
                                                              S, H);
    gemm_tdm<0, 0, 0, 0><<<mkgrid(BS, D), blk, 0, stream>>>(
        ao16, wo, infast, infast16, BS, D, D, nullptr, nullptr, nullptr,
        nullptr, nullptr);

    // --- CMS forward (original params) ---
    gemm_tdm<1, 1, 0, 0><<<mkgrid(BS, HID), blk, 0, stream>>>(
        infast16, w1f_in, nullptr, hbig16, BS, HID, D, b1f_in, nullptr,
        nullptr, nullptr, nullptr);
    gemm_tdm<1, 0, 0, 0><<<mkgrid(BS, D), blk, 0, stream>>>(
        hbig16, w2f_in, inslow, inslow16, BS, D, HID, b2f_in, nullptr, nullptr,
        nullptr, nullptr);
    gemm_tdm<1, 1, 0, 0><<<mkgrid(BS, HID), blk, 0, stream>>>(
        inslow16, w1s_in, nullptr, hbig16, BS, HID, D, b1s_in, nullptr,
        nullptr, nullptr, nullptr);
    gemm_tdm<1, 0, 0, 0><<<mkgrid(BS, D), blk, 0, stream>>>(
        hbig16, w2s_in, cms, nullptr, BS, D, HID, b2s_in, nullptr, nullptr,
        nullptr, nullptr);

    // --- TTT scans (sequential; params in d_out updated in place) ---
    auto run_ttt = [&](const float* lin, float* W1, float* B1, float* W2,
                       float* B2, int P) {
        const int nc = S / P, R = Bn * P;
        for (int c = 0; c < nc; ++c) {
            ttt_prep<<<1, 256, 0, stream>>>(lin, te, S, D, P, c * P, Bn, cic,
                                            cic16, tgt, rmask, scal);
            // z = ci@W1+b1 (store z f32), h = gelu(z) (bf16)
            gemm_tdm<1, 2, 0, 0><<<mkgrid(R, HID), blk, 0, stream>>>(
                cic16, W1, nullptr, hb16, R, HID, D, B1, zb, nullptr, nullptr,
                nullptr);
            // dpred = scal0*(h@W2+b2 - tgt)*mask (bf16)
            gemm_tdm<1, 0, 1, 0><<<mkgrid(R, D), blk, 0, stream>>>(
                hb16, W2, nullptr, dpred16, R, D, HID, B2, nullptr, tgt,
                rmask, scal);
            // dz = (dpred@W2^T) * dgelu(z) (bf16)
            gemm_tdm<0, 0, 2, 1><<<mkgrid(R, HID), blk, 0, stream>>>(
                dpred16, W2, nullptr, dz16, R, HID, D, nullptr, zb, nullptr,
                nullptr, nullptr);
            // W1 -= lr*flag*(ci^T @ dz) ; W2 -= lr*flag*(h^T @ dpred)
            gemm_wupd<<<mkgrid(D, HID), blk, 0, stream>>>(cic16, dz16, W1, D,
                                                          HID, R, scal, LR);
            gemm_wupd<<<mkgrid(HID, D), blk, 0, stream>>>(hb16, dpred16, W2,
                                                          HID, D, R, scal, LR);
            bias_update<<<dim3(HID / 256), blk, 0, stream>>>(B1, dz16, R, HID,
                                                             scal, LR);
            bias_update<<<dim3(D / 256), blk, 0, stream>>>(B2, dpred16, R, D,
                                                           scal, LR);
        }
    };
    run_ttt(infast, w1f, b1f, w2f, b2f, 16);   // fast level, 128 steps
    run_ttt(inslow, w1s, b1s, w2s, b2s, 64);   // slow level, 32 steps
}
